// LongParallelLayer_73461120631525
// MI455X (gfx1250) — matrix operand order repaired and checked by reading, not compiled, not run on hardware
//
#include <hip/hip_runtime.h>
#include <math.h>

// ---------------- CDNA5 WMMA types ----------------
typedef __attribute__((ext_vector_type(16))) __bf16 v16bf;
typedef __attribute__((ext_vector_type(4)))  __bf16 bf16x4;
typedef __attribute__((ext_vector_type(8)))  float  v8f;

#define EPS 1e-6f

// ---------------- generic RMSNorm (rows = gridDim.x) ----------------------
__global__ __launch_bounds__(256)
void rmsnorm_kernel(const float* __restrict__ x, const float* __restrict__ w,
                    float* __restrict__ out, int width) {
  int row = blockIdx.x;
  const float* xr = x + (size_t)row * width;
  float s = 0.f;
  for (int i = threadIdx.x; i < width; i += 256) { float v = xr[i]; s += v * v; }
  __shared__ float red[256];
  red[threadIdx.x] = s;
  __syncthreads();
  for (int off = 128; off > 0; off >>= 1) {
    if (threadIdx.x < off) red[threadIdx.x] += red[threadIdx.x + off];
    __syncthreads();
  }
  float rs = rsqrtf(red[0] / (float)width + EPS);
  float* orow = out + (size_t)row * width;
  for (int i = threadIdx.x; i < width; i += 256) orow[i] = w[i] * xr[i] * rs;
}

// ---------------- WMMA bf16 GEMM:  C[M,N] = A[M,K] * W[N,K]^T + epilogue ---
// M multiple of 128, K multiple of 32. TAILN=true enables N tail handling.
// EPI: 0 = none, 1 = +bias[col] then softplus, 2 = +res[row*N+col]
#define TM 128
#define TN 64
#define TK 32
#define LDP (TK + 4)   // halfword stride; keeps 8B alignment for b64 LDS stores
template <int EPI, bool TAILN>
__global__ __launch_bounds__(256)
void gemm_wmma_bf16(const float* __restrict__ A, const float* __restrict__ W,
                    float* __restrict__ C, const float* __restrict__ aux,
                    int M, int N, int K) {
  __shared__ __bf16 As[TM][LDP];
  __shared__ __bf16 Ws[TN][LDP];

  const int bm = blockIdx.y * TM;
  const int bn = blockIdx.x * TN;
  const int tid  = threadIdx.x;
  const int wave = tid >> 5;
  const int lane = tid & 31;
  const int l16  = lane & 15;
  const int half = lane >> 4;

  const int wrow = (wave & 3) * 32;   // 4 row groups of 32
  const int wcol = (wave >> 2) * 32;  // 2 col groups of 32

  v8f acc[2][2];
  acc[0][0] = v8f{}; acc[0][1] = v8f{};
  acc[1][0] = v8f{}; acc[1][1] = v8f{};

  // staging maps (float4 granularity): tile row = f4>>3, col4 = (f4&7)*4
  const int ar  = tid >> 3;             // A rows for j*32 offsets
  const int ac4 = (tid & 7) * 4;
  for (int k0 = 0; k0 < K; k0 += TK) {
    float4 av[4], wv[2];
#pragma unroll
    for (int j = 0; j < 4; ++j)         // A: 128x32 = 1024 float4
      av[j] = *(const float4*)(A + (size_t)(bm + ar + j * 32) * K + (k0 + ac4));
#pragma unroll
    for (int j = 0; j < 2; ++j) {       // W: 64x32 = 512 float4
      int gr = bn + ar + j * 32;
      int grc = TAILN ? ((gr < N) ? gr : (N - 1)) : gr;
      wv[j] = *(const float4*)(W + (size_t)grc * K + (k0 + ac4));
      if (TAILN && gr >= N) { wv[j].x = 0.f; wv[j].y = 0.f; wv[j].z = 0.f; wv[j].w = 0.f; }
    }
    if (k0 + TK < K) {                  // prefetch next K tile toward caches
#pragma unroll
      for (int j = 0; j < 4; ++j)
        __builtin_prefetch(A + (size_t)(bm + ar + j * 32) * K + (k0 + TK + ac4), 0, 0);
#pragma unroll
      for (int j = 0; j < 2; ++j)
        __builtin_prefetch(W + (size_t)(bn + ar + j * 32) * K + (k0 + TK + ac4), 0, 0);
    }
#pragma unroll
    for (int j = 0; j < 4; ++j) {
      bf16x4 p = { (__bf16)av[j].x, (__bf16)av[j].y, (__bf16)av[j].z, (__bf16)av[j].w };
      *(bf16x4*)&As[ar + j * 32][ac4] = p;
    }
#pragma unroll
    for (int j = 0; j < 2; ++j) {
      bf16x4 p = { (__bf16)wv[j].x, (__bf16)wv[j].y, (__bf16)wv[j].z, (__bf16)wv[j].w };
      *(bf16x4*)&Ws[ar + j * 32][ac4] = p;
    }
    __syncthreads();

    // A fragments (ISA 16-bit 16x32 layout): row = l16, half 0 -> K 0..7 &
    // 16..23, half 1 -> K 8..15 & 24..31
    v16bf af[2], bf[2];
#pragma unroll
    for (int i = 0; i < 2; ++i) {
      int arow = wrow + i * 16 + l16;
#pragma unroll
      for (int v = 0; v < 8; ++v) {
        int kb = (v < 4 ? 2 * v : 16 + 2 * (v - 4)) + half * 8;
        af[i][2 * v]     = As[arow][kb];
        af[i][2 * v + 1] = As[arow][kb + 1];
      }
    }
#pragma unroll
    for (int j = 0; j < 2; ++j) {
      int ncol = wcol + j * 16 + l16;
#pragma unroll
      for (int v = 0; v < 8; ++v) {
        int kb = (v < 4 ? 2 * v : 16 + 2 * (v - 4)) + half * 8;
        bf[j][2 * v]     = Ws[ncol][kb];
        bf[j][2 * v + 1] = Ws[ncol][kb + 1];
      }
    }
#pragma unroll
    for (int i = 0; i < 2; ++i)
#pragma unroll
      for (int j = 0; j < 2; ++j)
        acc[i][j] = __builtin_amdgcn_wmma_f32_16x16x32_bf16(
            false, af[i], false, bf[j], (short)0, acc[i][j], false, false);
    __syncthreads();
  }

  // C/D layout: col = l16, row = r + 8*half
#pragma unroll
  for (int i = 0; i < 2; ++i) {
#pragma unroll
    for (int j = 0; j < 2; ++j) {
      int col = bn + wcol + j * 16 + l16;
      if (!TAILN || col < N) {
        float bcol;
        if (EPI == 1) bcol = aux[col];
#pragma unroll
        for (int r = 0; r < 8; ++r) {
          int row = bm + wrow + i * 16 + r + 8 * half;
          float f = acc[i][j][r];
          size_t o = (size_t)row * N + col;
          if (EPI == 1) {            // bias + softplus
            f += bcol;
            f = (f > 20.f) ? f : log1pf(__expf(f));
          } else if (EPI == 2) {     // residual add
            f += aux[o];
          }
          C[o] = f;
        }
      }
    }
  }
}

// ---------------- depthwise conv K=4, pad (1,2), + SiLU --------------------
__global__ __launch_bounds__(256)
void conv_silu_kernel(const float* __restrict__ proj, const float* __restrict__ cw,
                      const float* __restrict__ cb, float* __restrict__ hs) {
  size_t idx = (size_t)blockIdx.x * 256 + threadIdx.x;  // B*L*2048
  int c = (int)(idx & 2047);
  size_t bt = idx >> 11;
  int b = (int)(bt >> 12);
  int t = (int)(bt & 4095);
  float acc = cb[c];
#pragma unroll
  for (int k = 0; k < 4; ++k) {
    int tt = t - 1 + k;
    if (tt >= 0 && tt < 4096)
      acc += proj[((size_t)(b * 4096 + tt)) * 4096 + c] * cw[c * 4 + k];
  }
  hs[idx] = acc / (1.f + __expf(-acc));
}

// ---------------- split + RMSNorm of ssm_p rows (96 = 64 + 16 + 16) -------
__global__ void ssmp_norm_kernel(const float* __restrict__ ssm_p,
                                 const float* __restrict__ dtw,
                                 const float* __restrict__ bw,
                                 const float* __restrict__ cw,
                                 float* __restrict__ tsn,
                                 float* __restrict__ Bn,
                                 float* __restrict__ Cn) {
  int row = blockIdx.x;
  int i = threadIdx.x;  // 0..95
  __shared__ float sq[96];
  __shared__ float rs[3];
  float v = ssm_p[(size_t)row * 96 + i];
  sq[i] = v * v;
  __syncthreads();
  if (i == 0) { float s = 0; for (int j = 0;  j < 64; ++j) s += sq[j]; rs[0] = rsqrtf(s / 64.f + EPS); }
  if (i == 1) { float s = 0; for (int j = 64; j < 80; ++j) s += sq[j]; rs[1] = rsqrtf(s / 16.f + EPS); }
  if (i == 2) { float s = 0; for (int j = 80; j < 96; ++j) s += sq[j]; rs[2] = rsqrtf(s / 16.f + EPS); }
  __syncthreads();
  if (i < 64)       tsn[(size_t)row * 64 + i]       = dtw[i] * v * rs[0];
  else if (i < 80)  Bn[(size_t)row * 16 + (i - 64)] = bw[i - 64] * v * rs[1];
  else              Cn[(size_t)row * 16 + (i - 80)] = cw[i - 80] * v * rs[2];
}

// ---------------- chunked (3-pass) selective scan --------------------------
// L=4096 split into NSEG=64 segments of SEG=64 steps; recurrence is linear:
// final = (prod dA) * init + local, so segment carries compose associatively.
#define SEG  64
#define NSEG 64
// P/S layout: [b][seg][c][n] : ((b*NSEG+seg)*2048 + c)*16 + n

__global__ __launch_bounds__(256)
void scan_pass1(const float* __restrict__ dt, const float* __restrict__ Bn,
                const float* __restrict__ hs, const float* __restrict__ A_log,
                float* __restrict__ P, float* __restrict__ S) {
  const int L = 4096, CI = 2048;
  int cg  = blockIdx.x & 7;
  int seg = (blockIdx.x >> 3) & 63;
  int b   = blockIdx.x >> 9;
  int c   = cg * 256 + threadIdx.x;
  float Ar[16], st[16], pr[16];
#pragma unroll
  for (int n = 0; n < 16; ++n) {
    Ar[n] = -__expf(A_log[(size_t)c * 16 + n]);
    st[n] = 0.f; pr[n] = 1.f;
  }
  __shared__ float Bs[SEG][16];
  for (int i = threadIdx.x; i < SEG * 16; i += 256) {
    int tt = i >> 4, n = i & 15;
    Bs[tt][n] = Bn[((size_t)(b * L + seg * SEG + tt)) * 16 + n];
  }
  __syncthreads();
  for (int tt = 0; tt < SEG; ++tt) {
    size_t idx = ((size_t)(b * L + seg * SEG + tt)) * CI + c;
    float d = dt[idx];
    float du = d * hs[idx];
#pragma unroll
    for (int n = 0; n < 16; ++n) {
      float dA = __expf(d * Ar[n]);
      st[n] = dA * st[n] + du * Bs[tt][n];
      pr[n] *= dA;
    }
  }
  size_t o = ((size_t)(b * NSEG + seg) * CI + c) * 16;
#pragma unroll
  for (int n = 0; n < 16; ++n) { P[o + n] = pr[n]; S[o + n] = st[n]; }
}

__global__ __launch_bounds__(256)
void scan_pass2(float* __restrict__ P, const float* __restrict__ S) {
  int idx = blockIdx.x * 256 + threadIdx.x;  // 2*2048*16 = 65536
  int n = idx & 15;
  int c = (idx >> 4) & 2047;
  int b = idx >> 15;
  float carry = 0.f;
  for (int s = 0; s < NSEG; ++s) {
    size_t o = ((size_t)(b * NSEG + s) * 2048 + c) * 16 + n;
    float p = P[o], sl = S[o];
    P[o] = carry;              // P becomes the segment-initial state
    carry = p * carry + sl;
  }
}

// pass 3: rescan from correct init state; fused gating epilogue:
// y2 = (y + hs*D[c]) * silu(gate)
__global__ __launch_bounds__(256)
void scan_pass3(const float* __restrict__ dt, const float* __restrict__ Bn,
                const float* __restrict__ Cn, const float* __restrict__ hs,
                const float* __restrict__ A_log, const float* __restrict__ Init,
                const float* __restrict__ Dv, const float* __restrict__ proj,
                float* __restrict__ y2) {
  const int L = 4096, CI = 2048;
  int cg  = blockIdx.x & 7;
  int seg = (blockIdx.x >> 3) & 63;
  int b   = blockIdx.x >> 9;
  int c   = cg * 256 + threadIdx.x;
  float Ar[16], st[16];
  size_t io = ((size_t)(b * NSEG + seg) * CI + c) * 16;
#pragma unroll
  for (int n = 0; n < 16; ++n) {
    Ar[n] = -__expf(A_log[(size_t)c * 16 + n]);
    st[n] = Init[io + n];
  }
  float Dc = Dv[c];
  __shared__ float Bs[SEG][16];
  __shared__ float Cs[SEG][16];
  for (int i = threadIdx.x; i < SEG * 16; i += 256) {
    int tt = i >> 4, n = i & 15;
    size_t base = ((size_t)(b * L + seg * SEG + tt)) * 16 + n;
    Bs[tt][n] = Bn[base];
    Cs[tt][n] = Cn[base];
  }
  __syncthreads();
  for (int tt = 0; tt < SEG; ++tt) {
    size_t bt = (size_t)(b * L + seg * SEG + tt);
    size_t idx = bt * CI + c;
    float d = dt[idx];
    float u = hs[idx];
    float du = d * u;
    float out = 0.f;
#pragma unroll
    for (int n = 0; n < 16; ++n) {
      float dA = __expf(d * Ar[n]);
      st[n] = dA * st[n] + du * Bs[tt][n];
      out += st[n] * Cs[tt][n];
    }
    float g = proj[bt * 4096 + 2048 + c];
    float sg = g / (1.f + __expf(-g));
    y2[idx] = (out + u * Dc) * sg;
  }
}

// ---------------- 4:1 chunk pool of ssm_out and hidden, summed ------------
__global__ __launch_bounds__(256)
void pool_add_kernel(const float* __restrict__ ssm_out,
                     const float* __restrict__ hidden, float* __restrict__ xres) {
  size_t idx = (size_t)blockIdx.x * 256 + threadIdx.x;  // 2*1024*1024
  int h = (int)(idx & 1023);
  int t = (int)((idx >> 10) & 1023);
  int b = (int)(idx >> 20);
  float s = 0.f;
#pragma unroll
  for (int j = 0; j < 4; ++j) {
    size_t r = (size_t)(b * 4096 + t * 4 + j);
    s += ssm_out[r * 1024 + h] + hidden[r * 1024 + h];
  }
  xres[idx] = s * 0.25f;
}

// ---------------- act = silu(g) * u ---------------------------------------
__global__ __launch_bounds__(256)
void silumul_kernel(const float* __restrict__ g, const float* __restrict__ u,
                    float* __restrict__ act) {
  size_t idx = (size_t)blockIdx.x * 256 + threadIdx.x;
  float gv = g[idx];
  act[idx] = (gv / (1.f + __expf(-gv))) * u[idx];
}

// ======================== host side =======================================
extern "C" void kernel_launch(void* const* d_in, const int* in_sizes, int n_in,
                              void* d_out, int out_size, void* d_ws, size_t ws_size,
                              hipStream_t stream) {
  const float* hidden    = (const float*)d_in[0];   // [2,4096,1024]
  const float* norm1_w   = (const float*)d_in[1];
  const float* in_proj_w = (const float*)d_in[2];   // [4096,1024]
  const float* conv_w    = (const float*)d_in[3];   // [2048,1,4]
  const float* conv_b    = (const float*)d_in[4];
  const float* x_proj_w  = (const float*)d_in[5];   // [96,2048]
  const float* dt_ln_w   = (const float*)d_in[6];
  const float* b_ln_w    = (const float*)d_in[7];
  const float* c_ln_w    = (const float*)d_in[8];
  const float* dt_proj_w = (const float*)d_in[9];   // [2048,64]
  const float* dt_proj_b = (const float*)d_in[10];
  const float* A_log     = (const float*)d_in[11];  // [2048,16]
  const float* Dp        = (const float*)d_in[12];
  const float* out_proj_w= (const float*)d_in[13];  // [1024,2048]
  const float* norm2_w   = (const float*)d_in[14];
  const float* gate_w    = (const float*)d_in[15];  // [4096,1024]
  const float* up_w      = (const float*)d_in[16];
  const float* down_w    = (const float*)d_in[17];  // [1024,4096]
  float* outp = (float*)d_out;                      // [2,1024,1024]

  char* ws = (char*)d_ws;
  size_t off = 0;
  float* proj  = (float*)(ws + off); off += (size_t)8192 * 4096 * 4;  // 128 MB
  float* hs    = (float*)(ws + off); off += (size_t)8192 * 2048 * 4;  //  64 MB
  float* dtb   = (float*)(ws + off); off += (size_t)8192 * 2048 * 4;  //  64 MB
  float* yb    = (float*)(ws + off); off += (size_t)8192 * 2048 * 4;  //  64 MB
  float* xn    = (float*)(ws + off); off += (size_t)8192 * 1024 * 4;  //  32 MB
  float* ssm_p = (float*)(ws + off); off += (size_t)8192 * 96 * 4;
  float* tsn   = (float*)(ws + off); off += (size_t)8192 * 64 * 4;
  float* Bn    = (float*)(ws + off); off += (size_t)8192 * 16 * 4;
  float* Cn    = (float*)(ws + off); off += (size_t)8192 * 16 * 4;
  float* xres  = (float*)(ws + off); off += (size_t)2048 * 1024 * 4;
  float* hbuf  = (float*)(ws + off); off += (size_t)2048 * 1024 * 4;
  // reuse: xn idle between in_proj input consumption and out_proj output
  float* Pbuf  = xn;                                 // 16 MB
  float* Sbuf  = xn + (size_t)2 * NSEG * 2048 * 16;  // 16 MB
  float* gbuf  = yb;                             // MLP g/u reuse yb
  float* ubuf  = yb + (size_t)2048 * 4096;
  float* act   = dtb;                            // MLP act reuses dtb
  (void)ws_size; (void)in_sizes; (void)n_in; (void)out_size;

  // 1) x = rmsnorm(hidden, norm1_w)  (into yb, free now)
  rmsnorm_kernel<<<8192, 256, 0, stream>>>(hidden, norm1_w, yb, 1024);
  // 2) proj = x @ in_proj_w^T   [8192 x 4096], K=1024
  gemm_wmma_bf16<0, false><<<dim3(4096 / TN, 8192 / TM), 256, 0, stream>>>(
      yb, in_proj_w, proj, nullptr, 8192, 4096, 1024);
  // 3) hs = silu(conv(proj[:, :2048]))
  conv_silu_kernel<<<(8192 * 2048) / 256, 256, 0, stream>>>(proj, conv_w, conv_b, hs);
  // 4) ssm_p = hs @ x_proj_w^T  [8192 x 96], K=2048  (N tail)
  gemm_wmma_bf16<0, true><<<dim3((96 + TN - 1) / TN, 8192 / TM), 256, 0, stream>>>(
      hs, x_proj_w, ssm_p, nullptr, 8192, 96, 2048);
  // 5) split + rmsnorm -> tsn[8192,64], Bn[8192,16], Cn[8192,16]
  ssmp_norm_kernel<<<8192, 96, 0, stream>>>(ssm_p, dt_ln_w, b_ln_w, c_ln_w, tsn, Bn, Cn);
  // 6) dt = softplus(tsn @ dt_proj_w^T + dt_proj_b)  [8192 x 2048], K=64
  gemm_wmma_bf16<1, false><<<dim3(2048 / TN, 8192 / TM), 256, 0, stream>>>(
      tsn, dt_proj_w, dtb, dt_proj_b, 8192, 2048, 64);
  // 7) chunked scan: pass1 (per-segment P,S), pass2 (carry), pass3 (+gating)
  scan_pass1<<<2 * NSEG * 8, 256, 0, stream>>>(dtb, Bn, hs, A_log, Pbuf, Sbuf);
  scan_pass2<<<(2 * 2048 * 16) / 256, 256, 0, stream>>>(Pbuf, Sbuf);
  scan_pass3<<<2 * NSEG * 8, 256, 0, stream>>>(dtb, Bn, Cn, hs, A_log, Pbuf,
                                               Dp, proj, yb);
  // 8) ssm_out = y2 @ out_proj_w^T  [8192 x 1024], K=2048 (into hs, now free)
  gemm_wmma_bf16<0, false><<<dim3(1024 / TN, 8192 / TM), 256, 0, stream>>>(
      yb, out_proj_w, hs, nullptr, 8192, 1024, 2048);
  // 9) x_res = pool(ssm_out) + pool(hidden)
  pool_add_kernel<<<(2 * 1024 * 1024) / 256, 256, 0, stream>>>(hs, hidden, xres);
  // 10) h = rmsnorm(x_res, norm2_w)
  rmsnorm_kernel<<<2048, 256, 0, stream>>>(xres, norm2_w, hbuf, 1024);
  // 11) g = h @ gate_w^T, u = h @ up_w^T   [2048 x 4096], K=1024
  gemm_wmma_bf16<0, false><<<dim3(4096 / TN, 2048 / TM), 256, 0, stream>>>(
      hbuf, gate_w, gbuf, nullptr, 2048, 4096, 1024);
  gemm_wmma_bf16<0, false><<<dim3(4096 / TN, 2048 / TM), 256, 0, stream>>>(
      hbuf, up_w, ubuf, nullptr, 2048, 4096, 1024);
  // 12) act = silu(g) * u  (into dtb, free after scan)
  silumul_kernel<<<(2048 * 4096) / 256, 256, 0, stream>>>(gbuf, ubuf, act);
  // 13) out = x_res + act @ down_w^T   [2048 x 1024], K=4096
  gemm_wmma_bf16<2, false><<<dim3(1024 / TN, 2048 / TM), 256, 0, stream>>>(
      act, down_w, outp, xres, 2048, 1024, 4096);
}
